// TwoExpertGatedImputer_64733747085894
// MI455X (gfx1250) — matrix operand order repaired and checked by reading, not compiled, not run on hardware
//
#include <hip/hip_runtime.h>
#include <hip/hip_bf16.h>
#include <math.h>

#define B_ 16
#define T_ 2048
#define N_ 64
static constexpr size_t TOT = (size_t)B_ * T_ * N_; // 2097152

typedef __attribute__((ext_vector_type(16))) _Float16 v16h;
typedef __attribute__((ext_vector_type(8)))  float    v8f;

__device__ __forceinline__ float gelu_t(float x) {
    float x3 = x * x * x;
    return 0.5f * x * (1.0f + tanhf(0.7978845608028654f * (x + 0.044715f * x3)));
}

#define WMMA_F16(A, B, C) \
    __builtin_amdgcn_wmma_f32_16x16x32_f16(false, (A), false, (B), (short)0, (C), false, false)

// async global -> LDS copy of 16 bytes (GV mode, tracked on ASYNCcnt)
__device__ __forceinline__ void async_load_b128(unsigned lds_addr, const void* gaddr) {
    asm volatile("global_load_async_to_lds_b128 %0, %1, off"
                 :: "v"(lds_addr), "v"(gaddr) : "memory");
}
// async LDS -> global copy of 4 bytes
__device__ __forceinline__ void async_store_b32(void* gaddr, unsigned lds_addr) {
    asm volatile("global_store_async_from_lds_b32 %0, %1, off"
                 :: "v"(gaddr), "v"(lds_addr) : "memory");
}
__device__ __forceinline__ void wait_async0() {
    asm volatile("s_wait_asynccnt 0" ::: "memory");
}

// ---------------------------------------------------------------------------
// K1: warm_start_fill as a parallel gather (expected lookback ~2 with 50% miss)
// ---------------------------------------------------------------------------
__global__ __launch_bounds__(256) void k_warm_fill(const float* __restrict__ x,
                                                   const float* __restrict__ mm,
                                                   float* __restrict__ seed) {
    size_t i = (size_t)blockIdx.x * 256 + threadIdx.x;
    if (i >= TOT) return;
    int t = (int)((i >> 6) % T_);
    float v = x[i];
    bool miss = (mm[i] != 0.0f);
    if (!miss) { seed[i] = v; return; }
    float fv = 0.0f; bool fav = false;
    for (int d = 1; d <= t; ++d) {                 // forward-fill: last obs <= t
        size_t j = i - (size_t)d * N_;
        if (mm[j] == 0.0f) { fv = x[j]; fav = true; break; }
    }
    float bv = 0.0f; bool bav = false;
    for (int d = 1; d < T_ - t; ++d) {             // backward-fill: first obs > t
        size_t j = i + (size_t)d * N_;
        if (mm[j] == 0.0f) { bv = x[j]; bav = true; break; }
    }
    float out;
    if (fav && bav)      out = 0.5f * (fv + bv);
    else if (fav)        out = fv;
    else if (bav)        out = bv;
    else                 out = 0.0f;
    seed[i] = out;
}

// ---------------------------------------------------------------------------
// K2: adjacency = softmax(relu(E E^T)); E is 64x32 (tiny, one block)
// ---------------------------------------------------------------------------
__global__ __launch_bounds__(64) void k_adjacency(const float* __restrict__ emb,
                                                  float* __restrict__ adj) {
    __shared__ float e[64][32];
    int tid = threadIdx.x;
    for (int i = tid; i < 64 * 32; i += 64) e[i >> 5][i & 31] = emb[i];
    __syncthreads();
    int r = tid;
    float sc[64];
    float mx = -1e30f;
    for (int c = 0; c < 64; ++c) {
        float s = 0.0f;
        #pragma unroll
        for (int h = 0; h < 32; ++h) s += e[r][h] * e[c][h];
        s = s > 0.0f ? s : 0.0f;
        sc[c] = s;
        mx = fmaxf(mx, s);
    }
    float sum = 0.0f;
    for (int c = 0; c < 64; ++c) { float ee = __expf(sc[c] - mx); sc[c] = ee; sum += ee; }
    float inv = 1.0f / sum;
    for (int c = 0; c < 64; ++c) adj[r * 64 + c] = sc[c] * inv;
}

// ---------------------------------------------------------------------------
// K3: x_freq = irfft(rfft(x_seed) * (re + i*im)) via radix-2 Stockham in LDS.
// 8 sequences per block (one wave each), 2 x 8 x 2048 complex ping-pong = 256KB
// dynamic LDS (CDNA5: up to 320KB per workgroup). Inverse scale folded into the
// last stage so writeback is a pure LDS->global async copy.
// ---------------------------------------------------------------------------
__global__ __launch_bounds__(256) void k_freq(const float* __restrict__ seed,
                                              const float* __restrict__ fre,
                                              const float* __restrict__ fim,
                                              float* __restrict__ xfreq) {
    extern __shared__ float2 lds[];
    float2* bufA = lds;
    float2* bufB = lds + 8 * 2048;
    const int b  = blockIdx.x >> 3;
    const int n0 = (blockIdx.x & 7) * 8;
    const int tid = threadIdx.x;

    const float* sp = seed + (size_t)b * T_ * N_ + n0;
    for (int i = tid; i < 2048 * 8; i += 256) {
        int t = i >> 3, j = i & 7;
        bufA[j * 2048 + t] = make_float2(sp[(size_t)t * N_ + j], 0.0f);
    }
    __syncthreads();

    const int wv = tid >> 5, lane = tid & 31;
    float2* s = bufA + wv * 2048;
    float2* d = bufB + wv * 2048;
    const float PI = 3.14159265358979323846f;

    // forward FFT (11 stages)
    for (int stage = 0; stage < 11; ++stage) {
        int m = 1 << stage, l = 1024 >> stage;
        for (int i = lane; i < 1024; i += 32) {
            int jj = i >> stage, k = i & (m - 1);
            float2 c0 = s[jj * m + k];
            float2 c1 = s[jj * m + k + 1024];
            float ang = -PI * (float)jj / (float)l;
            float sn, cs; __sincosf(ang, &sn, &cs);
            float2 dif = make_float2(c0.x - c1.x, c0.y - c1.y);
            d[2 * jj * m + k]     = make_float2(c0.x + c1.x, c0.y + c1.y);
            d[2 * jj * m + k + m] = make_float2(dif.x * cs - dif.y * sn,
                                                dif.x * sn + dif.y * cs);
        }
        __syncthreads();
        float2* t2 = s; s = d; d = t2;
    }

    // spectral filter; enforce Hermitian symmetry for a real inverse
    {
        int n = n0 + wv;
        for (int k = lane; k <= 1024; k += 32) {
            float2 X = s[k];
            float fr = fre[(size_t)k * N_ + n];
            float fi = fim[(size_t)k * N_ + n];
            float2 Y = make_float2(X.x * fr - X.y * fi, X.x * fi + X.y * fr);
            d[k] = Y;
            if (k > 0 && k < 1024) d[2048 - k] = make_float2(Y.x, -Y.y);
        }
        __syncthreads();
        float2* t2 = s; s = d; d = t2;
    }

    // inverse FFT (conjugate twiddles); 1/2048 folded into last stage
    for (int stage = 0; stage < 11; ++stage) {
        int m = 1 << stage, l = 1024 >> stage;
        float scl = (stage == 10) ? (1.0f / 2048.0f) : 1.0f;
        for (int i = lane; i < 1024; i += 32) {
            int jj = i >> stage, k = i & (m - 1);
            float2 c0 = s[jj * m + k];
            float2 c1 = s[jj * m + k + 1024];
            float ang = PI * (float)jj / (float)l;
            float sn, cs; __sincosf(ang, &sn, &cs);
            float2 dif = make_float2(c0.x - c1.x, c0.y - c1.y);
            d[2 * jj * m + k]     = make_float2((c0.x + c1.x) * scl,
                                                (c0.y + c1.y) * scl);
            d[2 * jj * m + k + m] = make_float2((dif.x * cs - dif.y * sn) * scl,
                                                (dif.x * sn + dif.y * cs) * scl);
        }
        __syncthreads();
        float2* t2 = s; s = d; d = t2;
    }

    // pure copy LDS -> global via async store engine (real part only)
    {
        int n = n0 + wv;
        float* op = xfreq + (size_t)b * T_ * N_ + n;
        for (int t = lane; t < 2048; t += 32) {
            unsigned la = (unsigned)(uintptr_t)(s + t); // address of s[t].x
            async_store_b32(op + (size_t)t * N_, la);
        }
        wait_async0();
    }
}

// ---------------------------------------------------------------------------
// K4: fused phi + GCN mix (WMMA) + 3-layer gate MLP (WMMA) + softmax/impute.
// Block = 256 threads (8 waves), tile = 16 (b,t) pairs x 64 nodes = 1024 rows.
// Tile inputs are staged global->LDS with async b128 copies (no VGPR bounce).
// ---------------------------------------------------------------------------
__global__ __launch_bounds__(256) void k_main(
    const float* __restrict__ xin,   const float* __restrict__ mmask,
    const int*   __restrict__ rate_id,
    const float* __restrict__ gcn_w1, const float* __restrict__ gcn_b1,
    const float* __restrict__ gcn_w2, const float* __restrict__ gcn_b2,
    const float* __restrict__ rate_table,
    const float* __restrict__ gw1, const float* __restrict__ gb1,
    const float* __restrict__ gw2, const float* __restrict__ gb2,
    const float* __restrict__ gw3, const float* __restrict__ gb3,
    const float* __restrict__ seed, const float* __restrict__ adj,
    const float* __restrict__ xfreq,
    float* __restrict__ xgcn_out, float* __restrict__ ximp_out,
    float* __restrict__ xcomp_out, float* __restrict__ gwt_out) {

    __shared__ float xs[16][64], mk[16][64], xf[16][64], xg[16][64];
    __shared__ _Float16 phi[16][64];
    __shared__ _Float16 adjB[64][64];     // [m][n] = adj[n][m]  (B for Phi @ A^T)
    __shared__ _Float16 w1h[32][64];      // gw1 zero-padded 22->32 in K
    __shared__ _Float16 w2h[64][64];
    __shared__ _Float16 w3h[64][16];      // gw3 zero-padded 2->16 in N
    __shared__ _Float16 rateh[64][16];
    __shared__ float b1s[64], b2s[64], b3s[2];
    __shared__ float cw1[32], cb1[32], cw2[32];
    __shared__ _Float16 scx[8][16][64];   // per-wave relayout scratch
    __shared__ float lg[8][16][2];

    const int tid = threadIdx.x;
    const size_t R = (size_t)blockIdx.x * 1024;

    // async-stage the three per-tile input planes straight into LDS
    {
        unsigned loff = (unsigned)tid * 16u;
        async_load_b128((unsigned)(uintptr_t)(&xs[0][0]) + loff, seed  + R + tid * 4);
        async_load_b128((unsigned)(uintptr_t)(&mk[0][0]) + loff, mmask + R + tid * 4);
        async_load_b128((unsigned)(uintptr_t)(&xf[0][0]) + loff, xfreq + R + tid * 4);
    }
    // meanwhile stage + convert the (wave-invariant) weights on the VALU path
    for (int i = tid; i < 32 * 64; i += 256) {
        int k = i >> 6, c = i & 63;
        w1h[k][c] = (_Float16)(k < 22 ? gw1[k * 64 + c] : 0.0f);
    }
    for (int i = tid; i < 64 * 64; i += 256) {
        int k = i >> 6, c = i & 63;
        w2h[k][c]  = (_Float16)gw2[i];
        adjB[k][c] = (_Float16)adj[c * 64 + k];
    }
    for (int i = tid; i < 64 * 16; i += 256) {
        int k = i >> 4, c = i & 15;
        w3h[k][c]   = (_Float16)(c < 2 ? gw3[k * 2 + c] : 0.0f);
        rateh[k][c] = (_Float16)rate_table[rate_id[k] * 16 + c];
    }
    if (tid < 64) { b1s[tid] = gb1[tid]; b2s[tid] = gb2[tid]; }
    if (tid < 32) { cw1[tid] = gcn_w1[tid]; cb1[tid] = gcn_b1[tid]; cw2[tid] = gcn_w2[tid]; }
    if (tid < 2)  b3s[tid] = gb3[tid];
    wait_async0();
    __syncthreads();

    // phi(s) = sum_h gelu(s*w1[h]+b1[h]) * w2[h]  — collapses GCN hidden dim
    for (int i = tid; i < 1024; i += 256) {
        float sv = xs[i >> 6][i & 63];
        float p = 0.0f;
        #pragma unroll
        for (int h = 0; h < 32; ++h) p += gelu_t(sv * cw1[h] + cb1[h]) * cw2[h];
        phi[i >> 6][i & 63] = (_Float16)p;
    }
    __syncthreads();

    const int wv = tid >> 5, lane = tid & 31;
    const int lrow = lane & 15, hlf = lane >> 4;
    const float b2v = gcn_b2[0];

    // GCN mix: C(16x64) = Phi(16x64) @ A^T(64x64); waves 0..3, one 16-col tile each
    if (wv < 4) {
        const int c = wv;
        v8f acc;
        #pragma unroll
        for (int v = 0; v < 8; ++v) acc[v] = 0.0f;
        #pragma unroll
        for (int kk = 0; kk < 2; ++kk) {
            v16h a, bb;
            #pragma unroll
            for (int v = 0; v < 8; ++v) {
                int k0 = 32 * kk + hlf * 16 + 2 * v;
                a[2 * v]     = phi[lrow][k0];
                a[2 * v + 1] = phi[lrow][k0 + 1];
                int kb = 32 * kk + hlf * 16 + 2 * v;
                bb[2 * v]     = adjB[kb][c * 16 + lrow];
                bb[2 * v + 1] = adjB[kb + 1][c * 16 + lrow];
            }
            acc = WMMA_F16(a, bb, acc);
        }
        #pragma unroll
        for (int v = 0; v < 8; ++v) {
            int rr = v + hlf * 8, cc = c * 16 + lrow;
            float val = acc[v] + b2v;
            xg[rr][cc] = val;
            xgcn_out[R + (size_t)rr * 64 + cc] = val;
        }
    }
    __syncthreads();

    // preload gate-MLP B matrices (wave-invariant across tiles)
    v16h B1[4], B2k[2][4], B3k[2];
    #pragma unroll
    for (int c = 0; c < 4; ++c) {
        #pragma unroll
        for (int v = 0; v < 8; ++v) {
            int kb = hlf * 16 + 2 * v;
            B1[c][2 * v]     = w1h[kb][c * 16 + lrow];
            B1[c][2 * v + 1] = w1h[kb + 1][c * 16 + lrow];
            #pragma unroll
            for (int kk = 0; kk < 2; ++kk) {
                B2k[kk][c][2 * v]     = w2h[32 * kk + kb][c * 16 + lrow];
                B2k[kk][c][2 * v + 1] = w2h[32 * kk + kb + 1][c * 16 + lrow];
            }
        }
    }
    #pragma unroll
    for (int kk = 0; kk < 2; ++kk)
        #pragma unroll
        for (int v = 0; v < 8; ++v) {
            int kb = 32 * kk + hlf * 16 + 2 * v;
            B3k[kk][2 * v]     = w3h[kb][lrow];
            B3k[kk][2 * v + 1] = w3h[kb + 1][lrow];
        }

    auto loadA64 = [&](v16h (&a)[2]) {
        #pragma unroll
        for (int kk = 0; kk < 2; ++kk)
            #pragma unroll
            for (int v = 0; v < 8; ++v) {
                int k0 = 32 * kk + hlf * 16 + 2 * v;
                a[kk][2 * v]     = scx[wv][lrow][k0];
                a[kk][2 * v + 1] = scx[wv][lrow][k0 + 1];
            }
    };

    // gate MLP: 8 waves x 8 tiles of 16 rows each (1024 rows per block)
    for (int it = 0; it < 8; ++it) {
        const int r0 = (wv * 8 + it) * 16;
        const int rowa = r0 + lrow;
        const int btl = rowa >> 6, nn = rowa & 63;
        float fx0 = xs[btl][nn], fx1 = mk[btl][nn];
        float fx2 = xg[btl][nn], fx3 = xf[btl][nn];
        float fx4 = fx2 - fx0, fx5 = fx3 - fx0;

        // layer 1 A (16x32, features padded 22->32)
        v16h a1;
        #pragma unroll
        for (int v = 0; v < 8; ++v) {
            int k0 = hlf * 16 + 2 * v;
            #pragma unroll
            for (int q = 0; q < 2; ++q) {
                int k = k0 + q;
                _Float16 hv;
                if      (k == 0) hv = (_Float16)fx0;
                else if (k == 1) hv = (_Float16)fx1;
                else if (k == 2) hv = (_Float16)fx2;
                else if (k == 3) hv = (_Float16)fx3;
                else if (k == 4) hv = (_Float16)fx4;
                else if (k == 5) hv = (_Float16)fx5;
                else if (k < 22) hv = rateh[nn][k - 6];
                else             hv = (_Float16)0.0f;
                a1[2 * v + q] = hv;
            }
        }
        #pragma unroll
        for (int c = 0; c < 4; ++c) {
            v8f cc; float bv = b1s[c * 16 + lrow];
            #pragma unroll
            for (int v = 0; v < 8; ++v) cc[v] = bv;
            cc = WMMA_F16(a1, B1[c], cc);
            #pragma unroll
            for (int v = 0; v < 8; ++v)
                scx[wv][v + hlf * 8][c * 16 + lrow] = (_Float16)gelu_t(cc[v]);
        }
        asm volatile("s_wait_dscnt 0" ::: "memory");   // intra-wave LDS RAW

        // layer 2
        v16h a2[2];
        loadA64(a2);
        #pragma unroll
        for (int c = 0; c < 4; ++c) {
            v8f cc; float bv = b2s[c * 16 + lrow];
            #pragma unroll
            for (int v = 0; v < 8; ++v) cc[v] = bv;
            cc = WMMA_F16(a2[0], B2k[0][c], cc);
            cc = WMMA_F16(a2[1], B2k[1][c], cc);
            #pragma unroll
            for (int v = 0; v < 8; ++v)
                scx[wv][v + hlf * 8][c * 16 + lrow] = (_Float16)gelu_t(cc[v]);
        }
        asm volatile("s_wait_dscnt 0" ::: "memory");

        // layer 3 (N padded 2->16)
        v16h a3[2];
        loadA64(a3);
        v8f c3;
        #pragma unroll
        for (int v = 0; v < 8; ++v) c3[v] = (lrow < 2) ? b3s[lrow] : 0.0f;
        c3 = WMMA_F16(a3[0], B3k[0], c3);
        c3 = WMMA_F16(a3[1], B3k[1], c3);
        if (lrow < 2) {
            #pragma unroll
            for (int v = 0; v < 8; ++v) lg[wv][v + hlf * 8][lrow] = c3[v];
        }
        asm volatile("s_wait_dscnt 0" ::: "memory");

        // softmax over 2 experts + impute + complete (lanes 0..15 -> 16 rows)
        if (lane < 16) {
            int row = lane;
            int ra = r0 + row, bl = ra >> 6, n2 = ra & 63;
            size_t gr = R + ra;
            float l0 = lg[wv][row][0], l1 = lg[wv][row][1];
            float mxl = fmaxf(l0, l1);
            float e0 = __expf(l0 - mxl), e1 = __expf(l1 - mxl);
            float w0 = e0 / (e0 + e1), w1v = 1.0f - w0;
            float xgv = xg[bl][n2], xfv = xf[bl][n2];
            float imp = w0 * xgv + w1v * xfv;
            float comp = (mk[bl][n2] != 0.0f) ? imp : xin[gr];
            gwt_out[2 * gr]     = w0;
            gwt_out[2 * gr + 1] = w1v;
            ximp_out[gr]  = imp;
            xcomp_out[gr] = comp;
        }
    }
}

// ---------------------------------------------------------------------------
extern "C" void kernel_launch(void* const* d_in, const int* in_sizes, int n_in,
                              void* d_out, int out_size, void* d_ws, size_t ws_size,
                              hipStream_t stream) {
    (void)in_sizes; (void)n_in; (void)out_size; (void)d_ws; (void)ws_size;
    const float* x_input    = (const float*)d_in[0];
    const float* mmask      = (const float*)d_in[1];
    const int*   rate_id    = (const int*)d_in[2];
    const float* node_emb   = (const float*)d_in[3];
    const float* gcn_w1     = (const float*)d_in[4];
    const float* gcn_b1     = (const float*)d_in[5];
    const float* gcn_w2     = (const float*)d_in[6];
    const float* gcn_b2     = (const float*)d_in[7];
    const float* freq_re    = (const float*)d_in[8];
    const float* freq_im    = (const float*)d_in[9];
    const float* rate_table = (const float*)d_in[10];
    const float* gw1 = (const float*)d_in[11]; const float* gb1 = (const float*)d_in[12];
    const float* gw2 = (const float*)d_in[13]; const float* gb2 = (const float*)d_in[14];
    const float* gw3 = (const float*)d_in[15]; const float* gb3 = (const float*)d_in[16];

    float* out    = (float*)d_out;
    float* o_seed = out;
    float* o_gcn  = out + TOT;
    float* o_freq = out + 2 * TOT;
    float* o_imp  = out + 3 * TOT;
    float* o_comp = out + 4 * TOT;
    float* o_gw   = out + 5 * TOT;
    float* o_adj  = out + 7 * TOT;

    k_warm_fill<<<(int)(TOT / 256), 256, 0, stream>>>(x_input, mmask, o_seed);
    k_adjacency<<<1, 64, 0, stream>>>(node_emb, o_adj);
    k_freq<<<(B_ * N_) / 8, 256, 2 * 8 * 2048 * sizeof(float2), stream>>>(
        o_seed, freq_re, freq_im, o_freq);
    k_main<<<(B_ * T_) / 16, 256, 0, stream>>>(
        x_input, mmask, rate_id, gcn_w1, gcn_b1, gcn_w2, gcn_b2, rate_table,
        gw1, gb1, gw2, gb2, gw3, gb3, o_seed, o_adj, o_freq,
        o_gcn, o_imp, o_comp, o_gw);
}
